// MHA_22290880266489
// MI455X (gfx1250) — matrix operand order repaired and checked by reading, not compiled, not run on hardware
//
#include <hip/hip_runtime.h>

#define B_  4
#define N_  1024
#define D_  1024
#define H_  16
#define DH_ 64
#define M_  (B_ * N_)          // 4096 rows in projection GEMMs
#define SMEM_LD 1032           // padded row stride for score strip
#define ALD 40                 // padded LDS row stride for staged tiles (32 + 8 bf16)

typedef __bf16 bf16;
typedef __attribute__((ext_vector_type(16))) __bf16 v16bf;
typedef __attribute__((ext_vector_type(8)))  __bf16 v8bf;
typedef __attribute__((ext_vector_type(4)))  __bf16 v4bf;
typedef __attribute__((ext_vector_type(8)))  float  v8f;
typedef __attribute__((ext_vector_type(4)))  unsigned int u32x4;
typedef __attribute__((ext_vector_type(8)))  unsigned int u32x8;

__device__ __forceinline__ v8f wmma_bf16(v16bf a, v16bf b, v8f c) {
  return __builtin_amdgcn_wmma_f32_16x16x32_bf16(false, a, false, b, (short)0, c,
                                                 false, false);
}

__device__ __forceinline__ void wait_tensor0() {
#if __has_builtin(__builtin_amdgcn_s_wait_tensorcnt)
  __builtin_amdgcn_s_wait_tensorcnt(0);
#else
  asm volatile("s_wait_tensorcnt 0x0" ::: "memory");
#endif
}

// Low 32 bits of a generic pointer to __shared__ are the LDS byte offset
// (flat-LDS aperture mapping).
__device__ __forceinline__ unsigned lds_off(const void* p) {
  return (unsigned)(unsigned long long)p;
}

// ---- Tensor Data Mover: 2D tile (tile_k x tile_rows) global -> LDS ----
// D# per ISA 8.3/8.4. data_size=2B. LDS padding: 4 DWORDs (8 bf16) inserted
// every 16 DWORDs (32 bf16 = one tile row) => LDS row stride = ALD = 40 bf16.
__device__ __forceinline__ void tdm_load_2d(const bf16* gsrc, unsigned lds_byte,
                                            unsigned tile_k, unsigned tile_rows,
                                            unsigned row_stride_elems) {
  unsigned long long ga = (unsigned long long)(const void*)gsrc;
  u32x4 g0;
  g0[0] = 1u;                                                  // count=1 (valid user D#)
  g0[1] = lds_byte;                                            // lds_addr
  g0[2] = (unsigned)ga;                                        // global_addr[31:0]
  g0[3] = (unsigned)((ga >> 32) & 0x01FFFFFFull) | (2u << 30); // addr[56:32] | type=2
  u32x8 g1;
  // data_size=1 (2B) | pad_enable | pad_interval=3 (16 DW) | pad_amount=3 (4 DW)
  g1[0] = (1u << 16) | (1u << 20) | (3u << 22) | (3u << 25);
  g1[1] = (tile_k & 0xFFFFu) << 16;                      // tensor_dim0[15:0]
  g1[2] = ((tile_k >> 16) & 0xFFFFu)                     // tensor_dim0[31:16]
        | ((tile_rows & 0xFFFFu) << 16);                 // tensor_dim1[15:0]
  g1[3] = ((tile_rows >> 16) & 0xFFFFu)                  // tensor_dim1[31:16]
        | ((tile_k & 0xFFFFu) << 16);                    // tile_dim0
  g1[4] = (tile_rows & 0xFFFFu);                         // tile_dim1 (tile_dim2=0)
  g1[5] = row_stride_elems;                              // tensor_dim0_stride[31:0]
  g1[6] = 0;
  g1[7] = 0;
  asm volatile("tensor_load_to_lds %0, %1" :: "s"(g0), "s"(g1) : "memory");
}

// ---- WMMA operand loads (ISA 7.12.2 lane layouts) ----
// A (16x32): lane L = row m0+(L&15); chunks K[k0+(L>>4)*8,+8) and +16.
__device__ __forceinline__ v16bf load_a_g(const bf16* __restrict__ p0, int ld,
                                          int m0, int k0, int lane) {
  const bf16* p = p0 + (size_t)(m0 + (lane & 15)) * ld + (k0 + ((lane >> 4) << 3));
  v8bf c0 = *(const v8bf*)p;
  v8bf c1 = *(const v8bf*)(p + 16);
  v16bf a;
#pragma unroll
  for (int i = 0; i < 8; ++i) { a[i] = c0[i]; a[i + 8] = c1[i]; }
  return a;
}
// B (32x16): lane L = column n0+(L&15) (stored as row of B^T, ld = K stride);
// chunks K[k0+(L>>4)*8,+8) and +16.
__device__ __forceinline__ v16bf load_b_g(const bf16* __restrict__ p0, int ld,
                                          int n0, int k0, int lane) {
  const bf16* p = p0 + (size_t)(n0 + (lane & 15)) * ld + (k0 + ((lane >> 4) << 3));
  v8bf c0 = *(const v8bf*)p;
  v8bf c1 = *(const v8bf*)(p + 16);
  v16bf b;
#pragma unroll
  for (int i = 0; i < 8; ++i) { b[i] = c0[i]; b[i + 8] = c1[i]; }
  return b;
}
// Same layouts out of TDM-staged LDS tiles (row stride ALD).
__device__ __forceinline__ v16bf load_a_lds(const bf16* t, int row0, int lane) {
  const bf16* p = t + (row0 + (lane & 15)) * ALD + ((lane >> 4) << 3);
  v8bf c0 = *(const v8bf*)p;
  v8bf c1 = *(const v8bf*)(p + 16);
  v16bf a;
#pragma unroll
  for (int i = 0; i < 8; ++i) { a[i] = c0[i]; a[i + 8] = c1[i]; }
  return a;
}
__device__ __forceinline__ v16bf load_b_lds(const bf16* t, int col0, int lane) {
  const bf16* p = t + (col0 + (lane & 15)) * ALD + ((lane >> 4) << 3);
  v8bf c0 = *(const v8bf*)p;
  v8bf c1 = *(const v8bf*)(p + 16);
  v16bf b;
#pragma unroll
  for (int i = 0; i < 8; ++i) { b[i] = c0[i]; b[i + 8] = c1[i]; }
  return b;
}

// ---------------- fp32 -> bf16 conversion ----------------
__global__ __launch_bounds__(256)
void cvt_f32_bf16(const float* __restrict__ s, bf16* __restrict__ d, int n) {
  int i = (blockIdx.x * blockDim.x + threadIdx.x) << 2;
  if (i >= n) return;
  float4 v = *(const float4*)(s + i);
  v4bf o;
  o[0] = (bf16)v.x; o[1] = (bf16)v.y; o[2] = (bf16)v.z; o[3] = (bf16)v.w;
  *(v4bf*)(d + i) = o;
}

// ---------------- projection GEMM: Y = X @ W^T + bias ----------------
// TDM-staged, double-buffered. Block = 256 thr (8 waves): block tile 128M x 64N,
// wave w owns rows w*16..w*16+15 and all 4 N-tiles. K-step 32.
// mode 0: bf16 -> [B,H,N,DH]; mode 1: bf16 -> [B,H,DH,N]; mode 2: f32 -> [M,D].
__global__ __launch_bounds__(256)
void gemm_proj(const bf16* __restrict__ X, const bf16* __restrict__ W,
               const float* __restrict__ bias, void* __restrict__ out, int mode) {
  __shared__ bf16 At[2][128][ALD];   // 20.0 KB
  __shared__ bf16 Bt[2][64][ALD];    // 10.0 KB
  const int lane = threadIdx.x & 31;
  const int wave = threadIdx.x >> 5;
  const int mblk = blockIdx.x * 128;
  const int n0 = blockIdx.y * 64;

  if (wave == 0) {
    tdm_load_2d(X + (size_t)mblk * D_, lds_off(&At[0][0][0]), 32, 128, D_);
    tdm_load_2d(W + (size_t)n0 * D_, lds_off(&Bt[0][0][0]), 32, 64, D_);
  }
  v8f acc[4] = {};
  wait_tensor0();
  __syncthreads();

  for (int ks = 0; ks < 32; ++ks) {
    const int cur = ks & 1;
    if (ks < 31 && wave == 0) {        // TDM prefetch next K-slab into other buffer
      const int k = (ks + 1) * 32;
      tdm_load_2d(X + (size_t)mblk * D_ + k, lds_off(&At[cur ^ 1][0][0]), 32, 128, D_);
      tdm_load_2d(W + (size_t)n0 * D_ + k, lds_off(&Bt[cur ^ 1][0][0]), 32, 64, D_);
    }
    // Gather ALL operands into distinct registers first so the backend can
    // clause the ds_load_b128s, wait dscnt once, then issue 4 WMMAs back-to-back.
    v16bf a  = load_a_lds(&At[cur][0][0], wave * 16, lane);
    v16bf b0 = load_b_lds(&Bt[cur][0][0],  0, lane);
    v16bf b1 = load_b_lds(&Bt[cur][0][0], 16, lane);
    v16bf b2 = load_b_lds(&Bt[cur][0][0], 32, lane);
    v16bf b3 = load_b_lds(&Bt[cur][0][0], 48, lane);
    acc[0] = wmma_bf16(a, b0, acc[0]);
    acc[1] = wmma_bf16(a, b1, acc[1]);
    acc[2] = wmma_bf16(a, b2, acc[2]);
    acc[3] = wmma_bf16(a, b3, acc[3]);
    wait_tensor0();                    // next buffers landed (wave 0)
    __syncthreads();                   // everyone done reading cur + buffers ready
  }

  const int m0 = mblk + wave * 16;
  const int hl = lane >> 4, nl = lane & 15;
#pragma unroll
  for (int j = 0; j < 4; ++j) {
#pragma unroll
    for (int r = 0; r < 8; ++r) {
      int m = m0 + r + hl * 8;
      int n = n0 + j * 16 + nl;
      float y = acc[j][r] + bias[n];
      if (mode == 2) {
        ((float*)out)[(size_t)m * D_ + n] = y;
      } else {
        int b  = m >> 10, row = m & (N_ - 1);
        int h  = n >> 6,  dh  = n & (DH_ - 1);
        if (mode == 0)
          ((bf16*)out)[(((size_t)(b * H_ + h) * N_) + row) * DH_ + dh] = (bf16)y;
        else
          ((bf16*)out)[(((size_t)(b * H_ + h) * DH_) + dh) * N_ + row] = (bf16)y;
      }
    }
  }
}

// -------- fused scores + mask + softmax + weights-write + P@V --------
// grid = (N/16, H, B); block = 128 thr (4 waves).
__global__ __launch_bounds__(128)
void attn_fused(const bf16* __restrict__ qh, const bf16* __restrict__ kh,
                const bf16* __restrict__ vT, const unsigned char* __restrict__ mask,
                float* __restrict__ wout, bf16* __restrict__ attn_out) {
  __shared__ float smem[16][SMEM_LD];
  __shared__ float red[16][8];
  __shared__ float rowmax[16];
  __shared__ float rowsum[16];

  const int lane = threadIdx.x & 31;
  const int wave = threadIdx.x >> 5;
  const int b = blockIdx.z, h = blockIdx.y;
  const int row0 = blockIdx.x * 16;

  const bf16* qbase = qh + (size_t)(b * H_ + h) * N_ * DH_;
  const bf16* kbase = kh + (size_t)(b * H_ + h) * N_ * DH_;
  const bf16* vbase = vT + (size_t)(b * H_ + h) * DH_ * N_;
  const unsigned char* mbase = mask + (size_t)(b * H_ + h) * N_ * N_;
  float* wbase = wout + (size_t)(b * H_ + h) * N_ * N_;

  v16bf a0 = load_a_g(qbase, DH_, row0, 0, lane);
  v16bf a1 = load_a_g(qbase, DH_, row0, 32, lane);

  const int hl = lane >> 4, nl = lane & 15;
  const int tbase = wave * 16;

  // ---- pass 0: S = (q k^T)/8 + mask, software-pipelined B loads
  v16bf b0 = load_b_g(kbase, DH_, tbase * 16, 0, lane);
  v16bf b1 = load_b_g(kbase, DH_, tbase * 16, 32, lane);
  for (int t = 0; t < 16; ++t) {
    const int n0 = (tbase + t) * 16;
    v16bf nb0, nb1;
    if (t < 15) {                              // prefetch next tile's operands
      nb0 = load_b_g(kbase, DH_, n0 + 16, 0, lane);
      nb1 = load_b_g(kbase, DH_, n0 + 16, 32, lane);
    }
    v8f c = {};
    c = wmma_bf16(a0, b0, c);
    c = wmma_bf16(a1, b1, c);
#pragma unroll
    for (int r = 0; r < 8; ++r) {
      int row = r + hl * 8;
      int col = n0 + nl;
      float s = c[r] * 0.125f;
      if (mbase[(size_t)(row0 + row) * N_ + col]) s = -1e10f;
      smem[row][col] = s;
    }
    if (t < 15) { b0 = nb0; b1 = nb1; }
  }
  __syncthreads();

  // ---- pass 1: row max
  const int rr = threadIdx.x & 15;
  const int seg = threadIdx.x >> 4;
  float mx = -3.402823466e38f;
  for (int c = seg * 128; c < seg * 128 + 128; ++c) mx = fmaxf(mx, smem[rr][c]);
  red[rr][seg] = mx;
  __syncthreads();
  if (threadIdx.x < 16) {
    float m = red[threadIdx.x][0];
#pragma unroll
    for (int i = 1; i < 8; ++i) m = fmaxf(m, red[threadIdx.x][i]);
    rowmax[threadIdx.x] = m;
  }
  __syncthreads();

  // ---- pass 2: exp + row sum
  const float rm = rowmax[rr];
  float sum = 0.f;
  for (int c = seg * 128; c < seg * 128 + 128; ++c) {
    float e = __expf(smem[rr][c] - rm);
    smem[rr][c] = e;
    sum += e;
  }
  red[rr][seg] = sum;
  __syncthreads();
  if (threadIdx.x < 16) {
    float s = 0.f;
#pragma unroll
    for (int i = 0; i < 8; ++i) s += red[threadIdx.x][i];
    rowsum[threadIdx.x] = s;
  }
  __syncthreads();

  // ---- pass 3: normalize + single coalesced weights write
  for (int i = 0; i < 128; ++i) {
    int idx = i * 128 + threadIdx.x;
    int row = idx >> 10, col = idx & (N_ - 1);
    float p = smem[row][col] * (1.0f / rowsum[row]);
    smem[row][col] = p;
    wbase[(size_t)(row0 + row) * N_ + col] = p;
  }
  __syncthreads();

  // ---- pass 4: attn = P @ V (wave w owns dh-tile w*16), pipelined B loads
  const int d0 = wave * 16;
  v8f c = {};
  v16bf bb = load_b_g(vbase, N_, d0, 0, lane);
  for (int k = 0; k < N_; k += 32) {
    v16bf nbb;
    if (k + 32 < N_) nbb = load_b_g(vbase, N_, d0, k + 32, lane);
    v16bf a;
    {
      int arow = lane & 15;
      int k0 = k + ((lane >> 4) << 3);
#pragma unroll
      for (int i = 0; i < 8; ++i) {
        a[i]     = (bf16)smem[arow][k0 + i];
        a[i + 8] = (bf16)smem[arow][k0 + 16 + i];
      }
    }
    c = wmma_bf16(a, bb, c);
    if (k + 32 < N_) bb = nbb;
  }
#pragma unroll
  for (int r = 0; r < 8; ++r) {
    int row = row0 + r + hl * 8;
    int dh = d0 + nl;
    attn_out[((size_t)(b * N_ + row)) * D_ + h * DH_ + dh] = (bf16)c[r];
  }
}

// ------------------------------ launcher ------------------------------
extern "C" void kernel_launch(void* const* d_in, const int* in_sizes, int n_in,
                              void* d_out, int out_size, void* d_ws, size_t ws_size,
                              hipStream_t stream) {
  const float* Q  = (const float*)d_in[0];
  const float* K  = (const float*)d_in[1];
  const float* V  = (const float*)d_in[2];
  const unsigned char* mask = (const unsigned char*)d_in[3];
  const float* Wq = (const float*)d_in[4];
  const float* bq = (const float*)d_in[5];
  const float* Wk = (const float*)d_in[6];
  const float* bk = (const float*)d_in[7];
  const float* Wv = (const float*)d_in[8];
  const float* bv = (const float*)d_in[9];
  const float* Wo = (const float*)d_in[10];
  const float* bo = (const float*)d_in[11];

  float* x_out = (float*)d_out;
  float* w_out = x_out + (size_t)B_ * N_ * D_;

  const size_t MN = (size_t)M_ * D_;
  const size_t DD = (size_t)D_ * D_;
  char* ws = (char*)d_ws;
  bf16* Qb   = (bf16*)ws; ws += MN * sizeof(bf16);
  bf16* Kb   = (bf16*)ws; ws += MN * sizeof(bf16);
  bf16* Vb   = (bf16*)ws; ws += MN * sizeof(bf16);
  bf16* Wqb  = (bf16*)ws; ws += DD * sizeof(bf16);
  bf16* Wkb  = (bf16*)ws; ws += DD * sizeof(bf16);
  bf16* Wvb  = (bf16*)ws; ws += DD * sizeof(bf16);
  bf16* Wob  = (bf16*)ws; ws += DD * sizeof(bf16);
  bf16* qh   = (bf16*)ws; ws += MN * sizeof(bf16);
  bf16* kh   = (bf16*)ws; ws += MN * sizeof(bf16);
  bf16* vT   = (bf16*)ws; ws += MN * sizeof(bf16);
  bf16* attc = (bf16*)ws; ws += MN * sizeof(bf16);

  const int thr = 256;
  cvt_f32_bf16<<<dim3((unsigned)((MN / 4 + thr - 1) / thr)), dim3(thr), 0, stream>>>(Q, Qb, (int)MN);
  cvt_f32_bf16<<<dim3((unsigned)((MN / 4 + thr - 1) / thr)), dim3(thr), 0, stream>>>(K, Kb, (int)MN);
  cvt_f32_bf16<<<dim3((unsigned)((MN / 4 + thr - 1) / thr)), dim3(thr), 0, stream>>>(V, Vb, (int)MN);
  cvt_f32_bf16<<<dim3((unsigned)((DD / 4 + thr - 1) / thr)), dim3(thr), 0, stream>>>(Wq, Wqb, (int)DD);
  cvt_f32_bf16<<<dim3((unsigned)((DD / 4 + thr - 1) / thr)), dim3(thr), 0, stream>>>(Wk, Wkb, (int)DD);
  cvt_f32_bf16<<<dim3((unsigned)((DD / 4 + thr - 1) / thr)), dim3(thr), 0, stream>>>(Wv, Wvb, (int)DD);
  cvt_f32_bf16<<<dim3((unsigned)((DD / 4 + thr - 1) / thr)), dim3(thr), 0, stream>>>(Wo, Wob, (int)DD);

  dim3 gg(M_ / 128, D_ / 64);
  gemm_proj<<<gg, dim3(256), 0, stream>>>(Qb, Wqb, bq, qh, 0);
  gemm_proj<<<gg, dim3(256), 0, stream>>>(Kb, Wkb, bk, kh, 0);
  gemm_proj<<<gg, dim3(256), 0, stream>>>(Vb, Wvb, bv, vT, 1);

  attn_fused<<<dim3(N_ / 16, H_, B_), dim3(128), 0, stream>>>(qh, kh, vT, mask,
                                                              w_out, attc);

  gemm_proj<<<gg, dim3(256), 0, stream>>>(attc, Wob, bo, x_out, 2);
}
